// WeightedHGTConv_88785563943718
// MI455X (gfx1250) — compile-verified, matrix operand order checked by reading, not run on hardware
//
#include <hip/hip_runtime.h>
#include <stdint.h>

// ---------------------------------------------------------------------------
// WeightedHGTConv for MI455X (gfx1250, wave32).
// QKV projection via v_wmma_f32_16x16x32_f16 with per-type row masking;
// edge softmax/aggregation via L2-resident b128 gathers + atomics.
// ---------------------------------------------------------------------------

typedef __attribute__((ext_vector_type(16))) _Float16 v16h;
typedef __attribute__((ext_vector_type(8)))  float    v8f;

union V16H { v16h v; uint32_t u[8]; uint4 q[2]; };
union F8   { v8f  v; float    f[8]; };

#define DMODEL 64
#define HEADS  8
#define NTYPES 4
#define PACKW_ELEMS (3*4*4*2*32*16)   // proj * type * ntile * kblock * lane * halves

// ---------------------------------------------------------------- utilities
__global__ void zero_u32_kernel(uint32_t* __restrict__ p, int n) {
  int i = blockIdx.x * blockDim.x + threadIdx.x;
  if (i < n) p[i] = 0u;
}

__global__ void cvt_f16_kernel(const float* __restrict__ x, _Float16* __restrict__ y, int n) {
  int i = blockIdx.x * blockDim.x + threadIdx.x;
  if (i < n) y[i] = (_Float16)x[i];
}

// Pack Wq/Wk/Wv into the per-lane B-operand layout of v_wmma_f32_16x16x32_f16.
// B fragment: lane L holds column n = (L&15); half i in 0..15 maps to
// k = kb*32 + (i<8 ? i : i+8) + (L>=16 ? 8 : 0)   (mirrors the ISA A layout).
__global__ void pack_w_kernel(const float* __restrict__ Wq, const float* __restrict__ Wk,
                              const float* __restrict__ Wv, _Float16* __restrict__ bpack) {
  int gid = blockIdx.x * blockDim.x + threadIdx.x;
  if (gid >= PACKW_ELEMS) return;
  int i    =  gid        & 15;
  int lane = (gid >> 4)  & 31;
  int kb   = (gid >> 9)  & 1;
  int nt   = (gid >> 10) & 3;
  int t    = (gid >> 12) & 3;
  int p    =  gid >> 14;
  int k = kb * 32 + (i < 8 ? i : i + 8) + ((lane >= 16) ? 8 : 0);
  int n = nt * 16 + (lane & 15);
  const float* W = (p == 0) ? Wq : (p == 1) ? Wk : Wv;
  bpack[gid] = (_Float16)W[((size_t)t * DMODEL + k) * DMODEL + n];
}

// ------------------------------------------------------------ QKV via WMMA
// One wave handles a 16-node tile: D = sum_t mask_t(A) x W_t (per-type row
// masking keeps EXEC all-1s and gives free output selection).
__global__ void qkv_wmma_kernel(const _Float16* __restrict__ x16,
                                const _Float16* __restrict__ bpack,
                                const int* __restrict__ ntype,
                                const float* __restrict__ bq,
                                const float* __restrict__ bk,
                                const float* __restrict__ bv,
                                float* __restrict__ Qo, float* __restrict__ Ko,
                                float* __restrict__ Vo, int N, int ntiles) {
  int wave = (int)((blockIdx.x * (size_t)blockDim.x + threadIdx.x) >> 5);
  int lane = threadIdx.x & 31;
  if (wave >= ntiles) return;                 // wave-uniform branch
  int base = wave * 16;
  int lrow = base + (lane & 15);
  int row  = (lrow < N) ? lrow : (N - 1);
  int rtype = (lrow < N) ? ntype[row] : -1;
  int lsel = (lane >> 4) * 8;                 // k-offset of this half-wave

  // A fragment: 16x32 f16 per kblock; lane holds row (lane&15),
  // k-chunks {lsel, lsel+16} as two 16B loads.
  V16H a[2];
  const _Float16* xr = x16 + (size_t)row * DMODEL;
#pragma unroll
  for (int kb = 0; kb < 2; ++kb) {
    a[kb].q[0] = *(const uint4*)(xr + kb * 32 + lsel);
    a[kb].q[1] = *(const uint4*)(xr + kb * 32 + lsel + 16);
  }

  int hi = lane >> 4;
  int col0 = lane & 15;
  for (int p = 0; p < 3; ++p) {
    F8 acc[4];
#pragma unroll
    for (int nt = 0; nt < 4; ++nt)
#pragma unroll
      for (int r = 0; r < 8; ++r) acc[nt].f[r] = 0.f;

    for (int t = 0; t < NTYPES; ++t) {
      unsigned msk = (rtype == t) ? 0xFFFFFFFFu : 0u;   // row mask (branchless)
      V16H am[2];
#pragma unroll
      for (int kb = 0; kb < 2; ++kb)
#pragma unroll
        for (int w = 0; w < 8; ++w) am[kb].u[w] = a[kb].u[w] & msk;
#pragma unroll
      for (int nt = 0; nt < 4; ++nt) {
#pragma unroll
        for (int kb = 0; kb < 2; ++kb) {
          V16H b;
          const _Float16* bp =
              bpack + ((size_t)((((p * 4 + t) * 4 + nt) * 2 + kb) * 32 + lane)) * 16;
          b.q[0] = *(const uint4*)(bp);
          b.q[1] = *(const uint4*)(bp + 8);
          acc[nt].v = __builtin_amdgcn_wmma_f32_16x16x32_f16(
              false, am[kb].v, false, b.v, (short)0, acc[nt].v, false, false);
        }
      }
    }
    // Epilogue: C/D layout — vgpr r, lanes0-15 -> row r, lanes16-31 -> row r+8.
    const float* bb = (p == 0) ? bq : (p == 1) ? bk : bv;
    float* op       = (p == 0) ? Qo : (p == 1) ? Ko : Vo;
#pragma unroll
    for (int r = 0; r < 8; ++r) {
      int rr = base + r + hi * 8;
      if (rr < N) {
        int tt = ntype[rr];
#pragma unroll
        for (int nt = 0; nt < 4; ++nt) {
          int col = nt * 16 + col0;
          op[(size_t)rr * DMODEL + col] = acc[nt].f[r] + bb[tt * DMODEL + col];
        }
      }
    }
  }
}

// ------------------------------------------------------------- edge pass A
// scores + segment max via order-preserving int keys. Q/K row gathers are
// explicit float4 (global_load_b128), L2-resident by design.
__global__ void edge_score_kernel(const int* __restrict__ ei, const int* __restrict__ etype,
                                  const int* __restrict__ esign, const float* __restrict__ edist,
                                  const float* __restrict__ Q, const float* __restrict__ K,
                                  const float* __restrict__ rel_q, const float* __restrict__ rel_k,
                                  const float* __restrict__ rel_bias, const float* __restrict__ sKn,
                                  const float* __restrict__ dist_alpha,
                                  const float* __restrict__ dist_tau,
                                  float* __restrict__ scores, uint32_t* __restrict__ mkey, int E) {
  __shared__ float s_rq[512], s_rk[512], s_rb[64], s_skn[64];
  for (int i = threadIdx.x; i < 512; i += blockDim.x) { s_rq[i] = rel_q[i]; s_rk[i] = rel_k[i]; }
  for (int i = threadIdx.x; i < 64;  i += blockDim.x) { s_rb[i] = rel_bias[i]; s_skn[i] = sKn[i]; }
  __syncthreads();
  int e = blockIdx.x * blockDim.x + threadIdx.x;
  if (e >= E) return;
  int src = ei[e], dst = ei[(size_t)E + e];
  int et = etype[e], sg = esign[e];
  float phi = dist_alpha[0] * expf(-edist[e] / (dist_tau[0] + 1e-9f));
  const float4* qr = (const float4*)(Q + (size_t)dst * DMODEL);   // L2-resident
  const float4* kr = (const float4*)(K + (size_t)src * DMODEL);
  const float* rq = s_rq + et * DMODEL;
  const float* rk = s_rk + et * DMODEL;
  const float rscale = 0.3535533905932738f;     // 1/sqrt(DK)
#pragma unroll
  for (int h = 0; h < HEADS; ++h) {
    float4 qa = qr[h * 2], qb = qr[h * 2 + 1];  // 8 q floats via 2x b128
    float4 ka = kr[h * 2], kb = kr[h * 2 + 1];
    float qv[8] = {qa.x, qa.y, qa.z, qa.w, qb.x, qb.y, qb.z, qb.w};
    float kv[8] = {ka.x, ka.y, ka.z, ka.w, kb.x, kb.y, kb.z, kb.w};
    float s = 0.f;
#pragma unroll
    for (int d = 0; d < 8; ++d) {
      int c = h * 8 + d;
      float sk = (sg == -1) ? -1.f : ((sg == 1) ? 1.f : s_skn[c]);
      s += (qv[d] * rq[c]) * (kv[d] * rk[c] * sk);
    }
    s = s * rscale + s_rb[et * 8 + h] + phi;
    scores[(size_t)e * 8 + h] = s;
    uint32_t b = __float_as_uint(s);
    uint32_t key = (b & 0x80000000u) ? ~b : (b | 0x80000000u);  // monotone key
    atomicMax(&mkey[(size_t)dst * 8 + h], key);
  }
}

// ------------------------------------------------------------- edge pass B
__global__ void edge_exp_kernel(const int* __restrict__ ei, const uint32_t* __restrict__ mkey,
                                float* __restrict__ scores, float* __restrict__ Z, int E) {
  int e = blockIdx.x * blockDim.x + threadIdx.x;
  if (e >= E) return;
  int dst = ei[(size_t)E + e];
#pragma unroll
  for (int h = 0; h < HEADS; ++h) {
    uint32_t key = mkey[(size_t)dst * 8 + h];
    float m = (key & 0x80000000u) ? __uint_as_float(key ^ 0x80000000u)
                                  : __uint_as_float(~key);
    float ex = expf(scores[(size_t)e * 8 + h] - m);
    scores[(size_t)e * 8 + h] = ex;               // cache ex for pass C
    atomicAdd(&Z[(size_t)dst * 8 + h], ex);
  }
}

// ------------------------------------------------------------- edge pass C
__global__ void edge_msg_kernel(const int* __restrict__ ei, const int* __restrict__ etype,
                                const int* __restrict__ esign, const float* __restrict__ V,
                                const float* __restrict__ rel_v, const float* __restrict__ sVn,
                                const float* __restrict__ scores, const float* __restrict__ Z,
                                float* __restrict__ out, int E) {
  __shared__ float s_rv[512], s_svn[64];
  for (int i = threadIdx.x; i < 512; i += blockDim.x) s_rv[i] = rel_v[i];
  for (int i = threadIdx.x; i < 64;  i += blockDim.x) s_svn[i] = sVn[i];
  __syncthreads();
  int e = blockIdx.x * blockDim.x + threadIdx.x;
  if (e >= E) return;
  int src = ei[e], dst = ei[(size_t)E + e];
  int et = etype[e], sg = esign[e];
  float w[HEADS];
#pragma unroll
  for (int h = 0; h < HEADS; ++h)
    w[h] = scores[(size_t)e * 8 + h] / (Z[(size_t)dst * 8 + h] + 1e-9f);
  const float4* vr = (const float4*)(V + (size_t)src * DMODEL);
  float* orow = out + (size_t)dst * DMODEL;
#pragma unroll
  for (int q = 0; q < 16; ++q) {                // 16x b128 gathers of V row
    float4 v4 = vr[q];
    float vv[4] = {v4.x, v4.y, v4.z, v4.w};
#pragma unroll
    for (int j = 0; j < 4; ++j) {
      int c = q * 4 + j;
      float sv = (sg == -1) ? -1.f : ((sg == 1) ? 1.f : s_svn[c]);
      atomicAdd(&orow[c], vv[j] * s_rv[et * DMODEL + c] * sv * w[c >> 3]);
    }
  }
}

// ----------------------------------------------------- skip + layernorm
// Two-pass (sum/sumsq then normalize) to avoid a 64-float live array.
__global__ void finalize_kernel(const float* __restrict__ xin, const int* __restrict__ ntype,
                                const float* __restrict__ skip, const float* __restrict__ gamma,
                                const float* __restrict__ beta, float* __restrict__ out, int N) {
  int n = blockIdx.x * blockDim.x + threadIdx.x;
  if (n >= N) return;
  int t = ntype[n];
  float alpha = 1.f / (1.f + expf(-skip[t]));
  const float4* xr = (const float4*)(xin + (size_t)n * DMODEL);
  float4* orow = (float4*)(out + (size_t)n * DMODEL);
  float sum = 0.f, sumsq = 0.f;
#pragma unroll
  for (int q = 0; q < 16; ++q) {
    float4 o4 = orow[q];
    float4 x4 = xr[q];
    float hv[4] = {alpha * o4.x + (1.f - alpha) * x4.x,
                   alpha * o4.y + (1.f - alpha) * x4.y,
                   alpha * o4.z + (1.f - alpha) * x4.z,
                   alpha * o4.w + (1.f - alpha) * x4.w};
#pragma unroll
    for (int j = 0; j < 4; ++j) { sum += hv[j]; sumsq += hv[j] * hv[j]; }
  }
  float mu = sum * (1.f / DMODEL);
  float var = sumsq * (1.f / DMODEL) - mu * mu;
  float inv = rsqrtf(var + 1e-5f);
#pragma unroll
  for (int q = 0; q < 16; ++q) {
    float4 o4 = orow[q];
    float4 x4 = xr[q];
    float4 r;
    float hv[4] = {alpha * o4.x + (1.f - alpha) * x4.x,
                   alpha * o4.y + (1.f - alpha) * x4.y,
                   alpha * o4.z + (1.f - alpha) * x4.z,
                   alpha * o4.w + (1.f - alpha) * x4.w};
    int c0 = q * 4;
    r.x = (hv[0] - mu) * inv * gamma[t * DMODEL + c0 + 0] + beta[t * DMODEL + c0 + 0];
    r.y = (hv[1] - mu) * inv * gamma[t * DMODEL + c0 + 1] + beta[t * DMODEL + c0 + 1];
    r.z = (hv[2] - mu) * inv * gamma[t * DMODEL + c0 + 2] + beta[t * DMODEL + c0 + 2];
    r.w = (hv[3] - mu) * inv * gamma[t * DMODEL + c0 + 3] + beta[t * DMODEL + c0 + 3];
    orow[q] = r;
  }
}

// --------------------------------------------------------------- launcher
extern "C" void kernel_launch(void* const* d_in, const int* in_sizes, int n_in,
                              void* d_out, int out_size, void* d_ws, size_t ws_size,
                              hipStream_t stream) {
  const float* node_inp  = (const float*)d_in[0];
  const int*   node_type = (const int*)  d_in[1];
  const int*   edge_idx  = (const int*)  d_in[2];
  const int*   edge_type = (const int*)  d_in[3];
  const int*   edge_sign = (const int*)  d_in[4];
  const float* edge_dist = (const float*)d_in[5];
  const float* Wq = (const float*)d_in[6];
  const float* bq = (const float*)d_in[7];
  const float* Wk = (const float*)d_in[8];
  const float* bk = (const float*)d_in[9];
  const float* Wv = (const float*)d_in[10];
  const float* bv = (const float*)d_in[11];
  const float* rel_q = (const float*)d_in[12];
  const float* rel_k = (const float*)d_in[13];
  const float* rel_v = (const float*)d_in[14];
  const float* sKn   = (const float*)d_in[15];
  const float* sVn   = (const float*)d_in[16];
  const float* rel_bias  = (const float*)d_in[17];
  const float* dist_alpha = (const float*)d_in[18];
  const float* dist_tau   = (const float*)d_in[19];
  const float* skip   = (const float*)d_in[20];
  const float* ln_g   = (const float*)d_in[21];
  const float* ln_b   = (const float*)d_in[22];

  const int N = in_sizes[1];     // node_type count
  const int E = in_sizes[3];     // edge_type count

  // workspace carve-out (256B aligned slices)
  char* ws = (char*)d_ws;
  size_t off = 0;
  auto carve = [&](size_t bytes) -> char* {
    char* p = ws + off;
    off = (off + bytes + 255) & ~(size_t)255;
    return p;
  };
  _Float16* x16   = (_Float16*)carve((size_t)N * DMODEL * sizeof(_Float16));
  _Float16* bpack = (_Float16*)carve((size_t)PACKW_ELEMS * sizeof(_Float16));
  float*    Qb    = (float*)   carve((size_t)N * DMODEL * sizeof(float));
  float*    Kb    = (float*)   carve((size_t)N * DMODEL * sizeof(float));
  float*    Vb    = (float*)   carve((size_t)N * DMODEL * sizeof(float));
  float*    sc    = (float*)   carve((size_t)E * HEADS * sizeof(float));
  uint32_t* mkey  = (uint32_t*)carve((size_t)N * HEADS * sizeof(uint32_t));
  float*    Zb    = (float*)   carve((size_t)N * HEADS * sizeof(float));
  float*    out   = (float*)d_out;

  const int B = 256;
  auto blks = [](long long n, int b) { return (int)((n + b - 1) / b); };

  // init: zero out accumulator, segment-max keys (0 == -inf key), Z
  zero_u32_kernel<<<blks((long long)N * DMODEL, B), B, 0, stream>>>((uint32_t*)out, N * DMODEL);
  zero_u32_kernel<<<blks((long long)N * HEADS, B), B, 0, stream>>>(mkey, N * HEADS);
  zero_u32_kernel<<<blks((long long)N * HEADS, B), B, 0, stream>>>((uint32_t*)Zb, N * HEADS);

  // prep: f16 inputs + WMMA-layout packed weights
  cvt_f16_kernel<<<blks((long long)N * DMODEL, B), B, 0, stream>>>(node_inp, x16, N * DMODEL);
  pack_w_kernel<<<blks(PACKW_ELEMS, B), B, 0, stream>>>(Wq, Wk, Wv, bpack);

  // QKV projection (one wave per 16-node tile, 8 waves/block)
  int ntiles = (N + 15) / 16;
  qkv_wmma_kernel<<<blks((long long)ntiles * 32, B), B, 0, stream>>>(
      x16, bpack, node_type, bq, bk, bv, Qb, Kb, Vb, N, ntiles);

  // edge phase: scores+max, exp+sum, weighted scatter
  edge_score_kernel<<<blks(E, B), B, 0, stream>>>(
      edge_idx, edge_type, edge_sign, edge_dist, Qb, Kb,
      rel_q, rel_k, rel_bias, sKn, dist_alpha, dist_tau, sc, mkey, E);
  edge_exp_kernel<<<blks(E, B), B, 0, stream>>>(edge_idx, mkey, sc, Zb, E);
  edge_msg_kernel<<<blks(E, B), B, 0, stream>>>(
      edge_idx, edge_type, edge_sign, Vb, rel_v, sVn, sc, Zb, out, E);

  // skip + layernorm (in place on d_out)
  finalize_kernel<<<blks(N, B), B, 0, stream>>>(node_inp, node_type, skip, ln_g, ln_b, out, N);
}
